// RadialAEVComputer_44092134260986
// MI455X (gfx1250) — matrix operand order repaired and checked by reading, not compiled, not run on hardware
//
#include <hip/hip_runtime.h>
#include <stdint.h>

typedef __attribute__((ext_vector_type(16))) _Float16 v16h;
typedef __attribute__((ext_vector_type(8)))  float    v8f;

#ifndef M_PI_F
#define M_PI_F 3.14159265358979323846f
#endif

namespace {
constexpr int   kN      = 256;   // atoms per molecule
constexpr int   kP      = 16;    // radial shifts
constexpr int   kS      = 4;     // species
constexpr float kRc     = 5.2f;
constexpr float kEta    = 16.0f;
constexpr float kLog2e  = 1.44269504088896340736f;
constexpr float kCe     = -kEta * kLog2e;          // -eta * log2(e)
constexpr int   kWaves  = 8;     // waves per block (wave32)
constexpr int   kChunks = kN / 32;
constexpr int   kOhPad  = kN + 8;  // row stride 528B = 132 dwords -> bank stride 4
}

__global__ __launch_bounds__(kWaves * 32)
void radial_aev_wmma(const float* __restrict__ dist,    // [B,256,256]
                     const int*  __restrict__ species,  // [B,256] (1-based ids)
                     float* __restrict__ out)           // [B,256,64]
{
    __shared__ int s_spec[kN];                               // async-staged species
    __shared__ __align__(16) _Float16 s_oh_t[kP][kOhPad];    // one-hot, [s][j]
    __shared__ __align__(16) float    s_d[kWaves][kN];       // d
    __shared__ __align__(16) float    s_A[kWaves][kN];       // ce*d^2 + log2(fc*mask)

    const int tid  = threadIdx.x;
    const int lane = tid & 31;
    const int wave = tid >> 5;

    const int b = blockIdx.x >> 5;                    // 32 blocks per batch entry
    const int i = ((blockIdx.x & 31) * kWaves) + wave;

    // ---- species row: async global -> LDS (ASYNCcnt path) ----
    {
        const unsigned ldsoff = (unsigned)(uintptr_t)&s_spec[tid];
        const unsigned voff   = (unsigned)tid * 4u;
        const unsigned long long sbase =
            (unsigned long long)(uintptr_t)(species + (size_t)b * kN);
        asm volatile("global_load_async_to_lds_b32 %0, %1, %2\n\t"
                     "s_wait_asynccnt 0"
                     :: "v"(ldsoff), "v"(voff), "s"(sbase)
                     : "memory");
    }
    // transposed one-hot f16 table for the B operand (exact in f16)
    {
        const int sp = s_spec[tid];
        #pragma unroll
        for (int s = 0; s < kP; ++s)
            s_oh_t[s][tid] = (sp == s + 1) ? (_Float16)1.0f : (_Float16)0.0f;
    }
    __syncthreads();

    const float* __restrict__ drow = dist + ((size_t)b * kN + i) * (size_t)kN;

    // ---- phase 1: fold distance+cutoff+mask into (d, A_j) for all 256 j ----
    #pragma unroll
    for (int c = 0; c < kChunks; ++c) {
        const int   j   = c * 32 + lane;
        const float d   = drow[j];
        const float fc  = 0.5f * __cosf(d * (M_PI_F / kRc)) + 0.5f;
        const bool  m   = (d < kRc) && (d != 0.0f);
        const float fcm = m ? fc : 0.0f;
        // A_j = ce*d^2 + log2(fcm);  log2(0) = -inf kills masked terms exactly
        s_d[wave][j] = d;
        s_A[wave][j] = fmaf(kCe * d, d, __log2f(fcm));
    }
    // intra-wave LDS RAW; compiler inserts s_wait_dscnt

    // ---- per-lane fragment roles ----
    const int   row = lane & 15;      // A row p / B col s / D col s
    const int   hi  = lane >> 4;      // 0: lanes 0-15, 1: lanes 16-31
    const float shf = 0.9f + 0.26875f * (float)row;
    const float k1  = -2.0f * kCe * shf;   // per-lane linear coeff
    const float k2  = kCe * shf * shf;     // per-lane constant

    v8f acc = {};

    #pragma unroll
    for (int c = 0; c < kChunks; ++c) {
        const int j0 = c * 32;
        const int jA = j0 + hi * 8;    // A K-slots: jA..jA+7 and jA+16..jA+23
        const int jB = j0 + hi * 8;    // B K-slots: jB..jB+7 and jB+16..jB+23

        // vectorized LDS reads: 8x b128 (d,A) + 2x b128 (one-hot)
        float dv[16], Av[16];
        *(float4*)&dv[0]  = *(const float4*)&s_d[wave][jA];
        *(float4*)&dv[4]  = *(const float4*)&s_d[wave][jA + 4];
        *(float4*)&dv[8]  = *(const float4*)&s_d[wave][jA + 16];
        *(float4*)&dv[12] = *(const float4*)&s_d[wave][jA + 20];
        *(float4*)&Av[0]  = *(const float4*)&s_A[wave][jA];
        *(float4*)&Av[4]  = *(const float4*)&s_A[wave][jA + 4];
        *(float4*)&Av[8]  = *(const float4*)&s_A[wave][jA + 16];
        *(float4*)&Av[12] = *(const float4*)&s_A[wave][jA + 20];

        union { uint4 q[2]; v16h h; } bf;
        bf.q[0] = *(const uint4*)&s_oh_t[row][jB];
        bf.q[1] = *(const uint4*)&s_oh_t[row][jB + 16];

        v16h afrag;
        #pragma unroll
        for (int t = 0; t < 16; ++t) {
            const float L = fmaf(dv[t], k1, Av[t]) + k2;
            afrag[t] = (_Float16)exp2f(L);        // v_exp_f32 is native base-2
        }

        acc = __builtin_amdgcn_wmma_f32_16x16x32_f16(
            false, afrag, false, bf.h, (short)0, acc, false, false);
    }

    // ---- D[p][s] -> out[b,i, s*16 + p]
    // lane 0-15: acc[r] = D[r][lane];  lane 16-31: acc[r] = D[r+8][lane-16]
    if (row < kS) {
        float* orow = out + (((size_t)b * kN + i) * (size_t)(kS * kP))
                          + row * kP + hi * 8;
        *(float4*)(orow + 0) = make_float4(acc[0], acc[1], acc[2], acc[3]);
        *(float4*)(orow + 4) = make_float4(acc[4], acc[5], acc[6], acc[7]);
    }
}

extern "C" void kernel_launch(void* const* d_in, const int* in_sizes, int n_in,
                              void* d_out, int out_size, void* d_ws, size_t ws_size,
                              hipStream_t stream) {
    const float* dist = (const float*)d_in[0];
    const int*   spec = (const int*)d_in[1];
    float*       out  = (float*)d_out;

    const int BN = in_sizes[1];        // B * 256
    const int B  = BN / kN;

    dim3 grid(B * (kN / kWaves));      // 8 atoms per block -> 2048 blocks
    dim3 block(kWaves * 32);
    radial_aev_wmma<<<grid, block, 0, stream>>>(dist, spec, out);
}
